// GTELayer_42588895708004
// MI455X (gfx1250) — matrix operand order repaired and checked by reading, not compiled, not run on hardware
//
#include <hip/hip_runtime.h>
#include <cstdint>
#include <cstddef>

// ---------------- problem constants ----------------
#define N_NODES 2048
#define H_HEADS 8
#define D_HEAD  16
#define HD_DIM  128
#define ED_DIM  64
#define FF_DIM  512
#define EG_EDGES 131072

typedef __attribute__((ext_vector_type(16))) _Float16 v16h;
typedef __attribute__((ext_vector_type(8)))  _Float16 v8h;
typedef __attribute__((ext_vector_type(8)))  float    v8f;

__device__ __forceinline__ v8f wmma_f32_16x16x32(v16h a, v16h b, v8f c) {
  // (neg_a, A, neg_b, B, c_mod, C, reuse_a, reuse_b)
  return __builtin_amdgcn_wmma_f32_16x16x32_f16(false, a, false, b, (short)0, c,
                                                false, false);
}

// Pack two contiguous 8xf16 runs into a WMMA operand register set.
// With natural element order, A-frag = runs K=[8h,8h+8) and K=[16+8h,16+8h+8);
// B-frag (from an n-major/transposed tile) = the same two runs along one column.
__device__ __forceinline__ v16h pack16(v8h lo, v8h hi) {
  union { v16h v; v8h h[2]; } u;
  u.h[0] = lo; u.h[1] = hi;
  return u.v;
}

__device__ __forceinline__ v8h cvt_v8h(float4 a, float4 b) {
  v8h r;
  r[0] = (_Float16)a.x; r[1] = (_Float16)a.y; r[2] = (_Float16)a.z; r[3] = (_Float16)a.w;
  r[4] = (_Float16)b.x; r[5] = (_Float16)b.y; r[6] = (_Float16)b.z; r[7] = (_Float16)b.w;
  return r;
}

__device__ __forceinline__ float gelu_exact(float x) {
  return 0.5f * x * (1.0f + erff(x * 0.70710678118654752f));
}

// Padded LDS row strides (f16 elems): multiple of 8 (16B-aligned b128 loads),
// dword-stride {20,36} mod 64 hits 16 disjoint 4-bank groups -> conflict-free.
#define S40 40
#define S72 72

// ---------------- GraphNorm (per-column over nodes) ----------------
__global__ __launch_bounds__(256) void graph_norm_kernel(
    const float* __restrict__ x, const float* __restrict__ w,
    const float* __restrict__ b, const float* __restrict__ a,
    float* __restrict__ y, int rows, int cols) {
  __shared__ float red[256];
  __shared__ float s_mu, s_rstd;
  const int c = blockIdx.x;
  const int tid = threadIdx.x;

  float s = 0.f;
  for (int r = tid; r < rows; r += 256) s += x[(size_t)r * cols + c];
  red[tid] = s; __syncthreads();
  for (int o = 128; o > 0; o >>= 1) { if (tid < o) red[tid] += red[tid + o]; __syncthreads(); }
  if (tid == 0) s_mu = red[0] / (float)rows;
  __syncthreads();

  const float am = a[c] * s_mu;
  float v = 0.f;
  for (int r = tid; r < rows; r += 256) {
    float t = x[(size_t)r * cols + c] - am;
    v += t * t;
  }
  red[tid] = v; __syncthreads();
  for (int o = 128; o > 0; o >>= 1) { if (tid < o) red[tid] += red[tid + o]; __syncthreads(); }
  if (tid == 0) s_rstd = rsqrtf(red[0] / (float)rows + 1e-5f);
  __syncthreads();

  const float ww = w[c], bb = b[c], rstd = s_rstd;
  for (int r = tid; r < rows; r += 256)
    y[(size_t)r * cols + c] = ww * (x[(size_t)r * cols + c] - am) * rstd + bb;
}

// ---------------- LayerNorm over 64-wide rows (one wave32 per row) --------
__global__ __launch_bounds__(256) void layer_norm64_kernel(
    const float* __restrict__ in, const float* __restrict__ w,
    const float* __restrict__ b, float* __restrict__ out, int rows) {
  const int wave = threadIdx.x >> 5, lane = threadIdx.x & 31;
  const int row = blockIdx.x * 8 + wave;
  if (row >= rows) return;
  const float* p = in + (size_t)row * 64;
  float v0 = p[lane], v1 = p[lane + 32];
  float s = v0 + v1;
  for (int o = 16; o > 0; o >>= 1) s += __shfl_xor(s, o, 32);
  const float mu = s * (1.f / 64.f);
  const float d0 = v0 - mu, d1 = v1 - mu;
  float q = d0 * d0 + d1 * d1;
  for (int o = 16; o > 0; o >>= 1) q += __shfl_xor(q, o, 32);
  const float rstd = rsqrtf(q * (1.f / 64.f) + 1e-5f);
  float* po = out + (size_t)row * 64;
  po[lane]      = w[lane]      * d0 * rstd + b[lane];
  po[lane + 32] = w[lane + 32] * d1 * rstd + b[lane + 32];
}

// ---------------- generic WMMA GEMM: C = act((A[+A2]) @ W + bias)[+R] -----
// 4 waves / 64x64 tile / K stepped by 32. A staged row-major, W staged
// TRANSPOSED (n-major) so every fragment is two ds_load_b128.
__global__ __launch_bounds__(128) void gemm_wmma_kernel(
    const float* __restrict__ A, const float* __restrict__ A2,
    const float* __restrict__ W, const float* __restrict__ bias,
    const float* __restrict__ R, float* __restrict__ C,
    int M, int K, int N, int act) {
  __shared__ _Float16 As[64 * S40];
  __shared__ _Float16 Bt[64 * S40];
  const int tid = threadIdx.x;
  const int wave = tid >> 5, lane = tid & 31;
  const int half = lane >> 4, l16 = lane & 15;
  const int m0 = blockIdx.x * 64;
  const int n0 = blockIdx.y * 64;

  v8f acc[4] = {};
  for (int k0 = 0; k0 < K; k0 += 32) {
    {  // A tile: 64x32, vector loads + b128 LDS stores
      const int row = tid >> 1, cb = (tid & 1) * 16;
      const float4* pa = (const float4*)(A + (size_t)(m0 + row) * K + k0 + cb);
      float4 q0 = pa[0], q1 = pa[1], q2 = pa[2], q3 = pa[3];
      if (A2) {
        const float4* pa2 = (const float4*)(A2 + (size_t)(m0 + row) * K + k0 + cb);
        float4 r0 = pa2[0], r1 = pa2[1], r2 = pa2[2], r3 = pa2[3];
        q0.x += r0.x; q0.y += r0.y; q0.z += r0.z; q0.w += r0.w;
        q1.x += r1.x; q1.y += r1.y; q1.z += r1.z; q1.w += r1.w;
        q2.x += r2.x; q2.y += r2.y; q2.z += r2.z; q2.w += r2.w;
        q3.x += r3.x; q3.y += r3.y; q3.z += r3.z; q3.w += r3.w;
      }
      *(v8h*)(As + row * S40 + cb)     = cvt_v8h(q0, q1);
      *(v8h*)(As + row * S40 + cb + 8) = cvt_v8h(q2, q3);
    }
    {  // B tile: read 32x64 row-major coalesced, store transposed
      const int row = tid >> 2, nb = (tid & 3) * 16;
      const float4* pb = (const float4*)(W + (size_t)(k0 + row) * N + n0 + nb);
      float tmp[16];
      *(float4*)(tmp + 0)  = pb[0];
      *(float4*)(tmp + 4)  = pb[1];
      *(float4*)(tmp + 8)  = pb[2];
      *(float4*)(tmp + 12) = pb[3];
      #pragma unroll
      for (int u = 0; u < 16; ++u)
        Bt[(nb + u) * S40 + row] = (_Float16)tmp[u];
    }
    __syncthreads();
    const _Float16* arow = As + (wave * 16 + l16) * S40;
    v16h af = pack16(*(const v8h*)(arow + 8 * half),
                     *(const v8h*)(arow + 16 + 8 * half));
    #pragma unroll
    for (int nt = 0; nt < 4; ++nt) {
      const _Float16* bcol = Bt + (nt * 16 + l16) * S40 + 8 * half;
      v16h bf = pack16(*(const v8h*)bcol, *(const v8h*)(bcol + 16));
      acc[nt] = wmma_f32_16x16x32(af, bf, acc[nt]);
    }
    __syncthreads();
  }
  #pragma unroll
  for (int nt = 0; nt < 4; ++nt) {
    #pragma unroll
    for (int r = 0; r < 8; ++r) {
      const int m = m0 + wave * 16 + r + half * 8;
      const int n = n0 + nt * 16 + l16;
      float v = acc[nt][r];
      if (bias) v += bias[n];
      if (act == 1) v = gelu_exact(v);
      if (R) v += R[(size_t)m * N + n];
      C[(size_t)m * N + n] = v;
    }
  }
}

// ---------------- attention scores: S[h][i][j] = (Q_i.K_j)/4*adj2 + relpos
__global__ __launch_bounds__(256) void attn_scores_kernel(
    const float* __restrict__ Q, const float* __restrict__ Km,
    const float* __restrict__ adj2, const float* __restrict__ relp,
    float* __restrict__ S) {
  const int wave = threadIdx.x >> 5, lane = threadIdx.x & 31;
  const int half = lane >> 4, l16 = lane & 15;
  int task = blockIdx.x * 8 + wave;            // H * 128 * 128 tiles
  const int jt = task & 127; task >>= 7;
  const int it = task & 127; const int h = task >> 7;

  const v8h zv = {(_Float16)0, (_Float16)0, (_Float16)0, (_Float16)0,
                  (_Float16)0, (_Float16)0, (_Float16)0, (_Float16)0};
  // A-frag: Q row, K(=d) padded 16->32 with zeros. Run0 fully < 16; run1 is zero pad.
  const float4* pq =
      (const float4*)(Q + (size_t)(it * 16 + l16) * HD_DIM + h * 16 + 8 * half);
  v16h af = pack16(cvt_v8h(pq[0], pq[1]), zv);
  // B-frag: K rows; each half supplies d=[8h,8h+8) in run0, run1 is pad.
  v8h klo = zv, khi = zv;
  {
    const float4* pk =
        (const float4*)(Km + (size_t)(jt * 16 + l16) * HD_DIM + h * 16 + 8 * half);
    klo = cvt_v8h(pk[0], pk[1]);
  }
  v16h bf = pack16(klo, khi);

  v8f acc = {};
  acc = wmma_f32_16x16x32(af, bf, acc);
  #pragma unroll
  for (int r = 0; r < 8; ++r) {
    const int m = it * 16 + r + half * 8;      // dst / query row i
    const int n = jt * 16 + l16;               // src / key row j
    const size_t ij = (size_t)m * N_NODES + n;
    const float s = acc[r] * 0.25f * adj2[ij] + relp[ij * H_HEADS + h];
    S[((size_t)h * N_NODES + m) * N_NODES + n] = s;
  }
}

// ---------------- pe = en @ Wpe + bpe (K=64, N=8: scalar) -----------------
__global__ __launch_bounds__(256) void pe_kernel(
    const float* __restrict__ en, const float* __restrict__ Wpe,
    const float* __restrict__ bpe, float* __restrict__ pe) {
  const int idx = blockIdx.x * 256 + threadIdx.x;  // EG*H
  const int h = idx & 7, g = idx >> 3;
  const float* row = en + (size_t)g * ED_DIM;
  float s = bpe[h];
  #pragma unroll
  for (int c = 0; c < ED_DIM; ++c) s += row[c] * Wpe[c * H_HEADS + h];
  pe[idx] = s;
}

// ------- gather sc = S[dst,src]; scatter S[dst,src] = sc + pe -------------
__global__ __launch_bounds__(256) void gather_scatter_kernel(
    float* __restrict__ S, const int* __restrict__ src,
    const int* __restrict__ dst, const float* __restrict__ pe,
    float* __restrict__ sc) {
  const int idx = blockIdx.x * 256 + threadIdx.x;  // EG*H
  const int h = idx & 7, g = idx >> 3;
  const size_t off = ((size_t)h * N_NODES + dst[g]) * N_NODES + src[g];
  const float v = S[off];
  sc[idx] = v;
  S[off] = v + pe[idx];                            // edges unique -> race free
}

// ---------------- eo = sc @ Wap + bap (K=8: scalar) -----------------------
__global__ __launch_bounds__(256) void attn_proj_kernel(
    const float* __restrict__ sc, const float* __restrict__ Wap,
    const float* __restrict__ bap, float* __restrict__ eo) {
  const int idx = blockIdx.x * 256 + threadIdx.x;  // EG*ED
  const int c = idx & 63, g = idx >> 6;
  const float* row = sc + (size_t)g * H_HEADS;
  float s = bap[c];
  #pragma unroll
  for (int h = 0; h < H_HEADS; ++h) s += row[h] * Wap[h * ED_DIM + c];
  eo[idx] = s;
}

// ---------------- clip(-5,5) + softmax over contiguous 2048-row -----------
__global__ __launch_bounds__(256) void softmax2048_kernel(float* __restrict__ S) {
  __shared__ float red[256];
  const int tid = threadIdx.x;
  float* row = S + (size_t)blockIdx.x * N_NODES;   // blockIdx = h*N + i
  float v[8], mx = -1e30f;
  #pragma unroll
  for (int t = 0; t < 8; ++t) {
    float s = row[t * 256 + tid];
    s = fminf(fmaxf(s, -5.0f), 5.0f);
    v[t] = s;
    mx = fmaxf(mx, s);
  }
  red[tid] = mx; __syncthreads();
  for (int o = 128; o > 0; o >>= 1) { if (tid < o) red[tid] = fmaxf(red[tid], red[tid + o]); __syncthreads(); }
  mx = red[0]; __syncthreads();
  float sum = 0.f;
  #pragma unroll
  for (int t = 0; t < 8; ++t) { v[t] = __expf(v[t] - mx); sum += v[t]; }
  red[tid] = sum; __syncthreads();
  for (int o = 128; o > 0; o >>= 1) { if (tid < o) red[tid] += red[tid + o]; __syncthreads(); }
  const float inv = 1.0f / red[0];
  #pragma unroll
  for (int t = 0; t < 8; ++t) row[t * 256 + tid] = v[t] * inv;
}

// ---------------- wV[i, h*16+d] = sum_j A_h[i,j] * V[j,h,d] (WMMA) --------
// 8 waves/block share one head: V panel staged transposed in LDS per k-step;
// A-fragments come straight from S rows as float4 loads (two 8-elem runs).
__global__ __launch_bounds__(256) void wv_kernel(
    const float* __restrict__ S, const float* __restrict__ V,
    float* __restrict__ wV) {
  __shared__ _Float16 Vt[16 * S40];
  const int tid = threadIdx.x;
  const int wave = tid >> 5, lane = tid & 31;
  const int half = lane >> 4, l16 = lane & 15;
  const int b = blockIdx.x;                    // 128 blocks
  const int h = b >> 4;
  const int it = (b & 15) * 8 + wave;
  const size_t srow = ((size_t)h * N_NODES + it * 16 + l16) * N_NODES;

  v8f acc = {};
  for (int ks = 0; ks < N_NODES / 32; ++ks) {
    if (tid < 128) {                           // stage 32x16 V panel transposed
      const int j = tid >> 2, d4 = (tid & 3) * 4;
      float4 vv = *(const float4*)(V + (size_t)(ks * 32 + j) * HD_DIM + h * 16 + d4);
      Vt[(d4 + 0) * S40 + j] = (_Float16)vv.x;
      Vt[(d4 + 1) * S40 + j] = (_Float16)vv.y;
      Vt[(d4 + 2) * S40 + j] = (_Float16)vv.z;
      Vt[(d4 + 3) * S40 + j] = (_Float16)vv.w;
    }
    __syncthreads();
    const float* sp = S + srow + ks * 32 + 8 * half;
    v16h af = pack16(cvt_v8h(((const float4*)sp)[0], ((const float4*)sp)[1]),
                     cvt_v8h(((const float4*)(sp + 16))[0], ((const float4*)(sp + 16))[1]));
    const _Float16* vc = Vt + l16 * S40 + 8 * half;
    v16h bf = pack16(*(const v8h*)vc, *(const v8h*)(vc + 16));
    acc = wmma_f32_16x16x32(af, bf, acc);
    __syncthreads();
  }
  #pragma unroll
  for (int r = 0; r < 8; ++r) {
    const int m = it * 16 + r + half * 8;
    wV[(size_t)m * HD_DIM + h * 16 + l16] = acc[r];
  }
}

// ---------------- out = a + b ---------------------------------------------
__global__ __launch_bounds__(256) void add2_kernel(
    const float* __restrict__ a, const float* __restrict__ b,
    float* __restrict__ out, int n) {
  const int i = blockIdx.x * 256 + threadIdx.x;
  if (i < n) out[i] = a[i] + b[i];
}

// ------- fused edge FFN: out = e2 + (gelu(ln(e2)@We1+b1))@We2+b2 ----------
// 64-row tile; hidden(512) in 8 chunks of 64 kept in LDS f16; output accums
// stay in VGPRs. W1/W2 chunks staged transposed so all fragments are b128.
__global__ __launch_bounds__(128) void e_ffn_kernel(
    const float* __restrict__ e2ln, const float* __restrict__ e2,
    const float* __restrict__ W1, const float* __restrict__ b1,
    const float* __restrict__ W2, const float* __restrict__ b2,
    float* __restrict__ out) {
  __shared__ _Float16 Ash[64 * S72];   // input tile (row-major)
  __shared__ _Float16 Hs[64 * S72];    // hidden chunk (row-major)
  __shared__ _Float16 W1t[64 * S72];   // W1 chunk, transposed (n-major)
  __shared__ _Float16 W2t[64 * S72];   // W2 chunk, transposed (n-major)
  const int tid = threadIdx.x;
  const int wave = tid >> 5, lane = tid & 31;
  const int half = lane >> 4, l16 = lane & 15;
  const size_t r0 = (size_t)blockIdx.x * 64;

  {  // stage input tile
    const int row = tid >> 1, cb = (tid & 1) * 32;
    const float4* sp = (const float4*)(e2ln + (r0 + row) * ED_DIM + cb);
    #pragma unroll
    for (int q = 0; q < 4; ++q)
      *(v8h*)(Ash + row * S72 + cb + q * 8) = cvt_v8h(sp[2 * q], sp[2 * q + 1]);
  }
  __syncthreads();

  v16h a1[2];
  const _Float16* arow = Ash + (wave * 16 + l16) * S72;
  #pragma unroll
  for (int ks = 0; ks < 2; ++ks)
    a1[ks] = pack16(*(const v8h*)(arow + ks * 32 + 8 * half),
                    *(const v8h*)(arow + ks * 32 + 16 + 8 * half));

  v8f oacc[4] = {};
  for (int hc = 0; hc < 8; ++hc) {
    // stage W1[:, hc*64+n] and W2[hc*64+k, :] transposed (coalesced reads)
    #pragma unroll
    for (int itr = 0; itr < 8; ++itr) {
      const int e4 = tid + 128 * itr;          // 1024 float4 per array
      const int n = (e4 * 4) & 63;
      const int k = (e4 * 4) >> 6;
      float4 w1v = *(const float4*)(W1 + (size_t)k * FF_DIM + hc * 64 + n);
      W1t[(n + 0) * S72 + k] = (_Float16)w1v.x;
      W1t[(n + 1) * S72 + k] = (_Float16)w1v.y;
      W1t[(n + 2) * S72 + k] = (_Float16)w1v.z;
      W1t[(n + 3) * S72 + k] = (_Float16)w1v.w;
      float4 w2v = *(const float4*)(W2 + (size_t)(hc * 64 + k) * ED_DIM + n);
      W2t[(n + 0) * S72 + k] = (_Float16)w2v.x;
      W2t[(n + 1) * S72 + k] = (_Float16)w2v.y;
      W2t[(n + 2) * S72 + k] = (_Float16)w2v.z;
      W2t[(n + 3) * S72 + k] = (_Float16)w2v.w;
    }
    __syncthreads();
    // GEMM1: hidden chunk = gelu(Ash @ W1chunk + b1)
    #pragma unroll
    for (int nt = 0; nt < 4; ++nt) {
      v8f acc = {};
      const _Float16* bcol = W1t + (nt * 16 + l16) * S72 + 8 * half;
      #pragma unroll
      for (int ks = 0; ks < 2; ++ks) {
        v16h bf = pack16(*(const v8h*)(bcol + ks * 32),
                         *(const v8h*)(bcol + ks * 32 + 16));
        acc = wmma_f32_16x16x32(a1[ks], bf, acc);
      }
      const float bb = b1[hc * 64 + nt * 16 + l16];
      #pragma unroll
      for (int r = 0; r < 8; ++r)
        Hs[(wave * 16 + r + half * 8) * S72 + nt * 16 + l16] =
            (_Float16)gelu_exact(acc[r] + bb);
    }
    __syncthreads();
    // GEMM2 partial: oacc += hidden_chunk @ W2chunk
    const _Float16* hrow = Hs + (wave * 16 + l16) * S72;
    #pragma unroll
    for (int ks = 0; ks < 2; ++ks) {
      v16h af = pack16(*(const v8h*)(hrow + ks * 32 + 8 * half),
                       *(const v8h*)(hrow + ks * 32 + 16 + 8 * half));
      #pragma unroll
      for (int nt = 0; nt < 4; ++nt) {
        const _Float16* bcol = W2t + (nt * 16 + l16) * S72 + ks * 32 + 8 * half;
        v16h bf = pack16(*(const v8h*)bcol, *(const v8h*)(bcol + 16));
        oacc[nt] = wmma_f32_16x16x32(af, bf, oacc[nt]);
      }
    }
    __syncthreads();
  }
  #pragma unroll
  for (int nt = 0; nt < 4; ++nt) {
    #pragma unroll
    for (int r = 0; r < 8; ++r) {
      const int m = wave * 16 + r + half * 8;
      const int n = nt * 16 + l16;
      out[(r0 + m) * ED_DIM + n] = oacc[nt][r] + b2[n] + e2[(r0 + m) * ED_DIM + n];
    }
  }
}

// ===========================================================================
extern "C" void kernel_launch(void* const* d_in, const int* in_sizes, int n_in,
                              void* d_out, int out_size, void* d_ws, size_t ws_size,
                              hipStream_t stream) {
  (void)in_sizes; (void)n_in; (void)out_size; (void)ws_size;

  const float* x     = (const float*)d_in[0];
  const float* e     = (const float*)d_in[1];
  const float* adj2  = (const float*)d_in[2];
  const float* relp  = (const float*)d_in[3];
  const int*   src   = (const int*)d_in[4];
  const int*   dst   = (const int*)d_in[5];
  const float* gn1_w = (const float*)d_in[6];
  const float* gn1_b = (const float*)d_in[7];
  const float* gn1_a = (const float*)d_in[8];
  const float* ln1ew = (const float*)d_in[9];
  const float* ln1eb = (const float*)d_in[10];
  const float* Wq    = (const float*)d_in[11];
  const float* bq    = (const float*)d_in[12];
  const float* Wk    = (const float*)d_in[13];
  const float* bk    = (const float*)d_in[14];
  const float* Wv    = (const float*)d_in[15];
  const float* bv    = (const float*)d_in[16];
  const float* Wpe   = (const float*)d_in[17];
  const float* bpe   = (const float*)d_in[18];
  const float* Wap   = (const float*)d_in[19];
  const float* bap   = (const float*)d_in[20];
  const float* Wo    = (const float*)d_in[21];
  const float* bo    = (const float*)d_in[22];
  const float* Woe   = (const float*)d_in[23];
  const float* boe   = (const float*)d_in[24];
  const float* gn2_w = (const float*)d_in[25];
  const float* gn2_b = (const float*)d_in[26];
  const float* gn2_a = (const float*)d_in[27];
  const float* ln2ew = (const float*)d_in[28];
  const float* ln2eb = (const float*)d_in[29];
  const float* Wh1   = (const float*)d_in[30];
  const float* bh1   = (const float*)d_in[31];
  const float* Wh2   = (const float*)d_in[32];
  const float* bh2   = (const float*)d_in[33];
  const float* We1   = (const float*)d_in[34];
  const float* be1   = (const float*)d_in[35];
  const float* We2   = (const float*)d_in[36];
  const float* be2   = (const float*)d_in[37];

  float* outx = (float*)d_out;                      // x2: N*HD
  float* oute = outx + (size_t)N_NODES * HD_DIM;    // e2: EG*ED

  // workspace layout (floats)
  float* ws   = (float*)d_ws;
  float* y    = ws;                                  size_t o = (size_t)N_NODES * HD_DIM;
  float* Qm   = ws + o;  o += (size_t)N_NODES * HD_DIM;
  float* Km   = ws + o;  o += (size_t)N_NODES * HD_DIM;
  float* Vm   = ws + o;  o += (size_t)N_NODES * HD_DIM;
  float* en   = ws + o;  o += (size_t)EG_EDGES * ED_DIM;
  float* pe   = ws + o;  o += (size_t)EG_EDGES * H_HEADS;
  float* sc   = ws + o;  o += (size_t)EG_EDGES * H_HEADS;
  float* eo   = ws + o;  o += (size_t)EG_EDGES * ED_DIM;
  float* eo2  = ws + o;  o += (size_t)EG_EDGES * ED_DIM;
  float* S    = ws + o;  o += (size_t)N_NODES * N_NODES * H_HEADS;
  float* wV   = ws + o;  o += (size_t)N_NODES * HD_DIM;
  float* ho   = ws + o;  o += (size_t)N_NODES * HD_DIM;
  float* x2g  = ws + o;  o += (size_t)N_NODES * HD_DIM;
  float* hh   = ws + o;  o += (size_t)N_NODES * FF_DIM;
  float* e2   = ws + o;  o += (size_t)EG_EDGES * ED_DIM;
  float* e2ln = ws + o;

  // 1. norms
  graph_norm_kernel<<<HD_DIM, 256, 0, stream>>>(x, gn1_w, gn1_b, gn1_a, y, N_NODES, HD_DIM);
  layer_norm64_kernel<<<EG_EDGES / 8, 256, 0, stream>>>(e, ln1ew, ln1eb, en, EG_EDGES);

  // 2. Q/K/V projections (WMMA)
  dim3 gqkv(N_NODES / 64, HD_DIM / 64);
  gemm_wmma_kernel<<<gqkv, 128, 0, stream>>>(y, nullptr, Wq, bq, nullptr, Qm, N_NODES, HD_DIM, HD_DIM, 0);
  gemm_wmma_kernel<<<gqkv, 128, 0, stream>>>(y, nullptr, Wk, bk, nullptr, Km, N_NODES, HD_DIM, HD_DIM, 0);
  gemm_wmma_kernel<<<gqkv, 128, 0, stream>>>(y, nullptr, Wv, bv, nullptr, Vm, N_NODES, HD_DIM, HD_DIM, 0);

  // 3. edge bias projection
  pe_kernel<<<EG_EDGES * H_HEADS / 256, 256, 0, stream>>>(en, Wpe, bpe, pe);

  // 4. dense attention scores (WMMA) into S[h][i][j]
  attn_scores_kernel<<<(H_HEADS * 128 * 128) / 8, 256, 0, stream>>>(Qm, Km, adj2, relp, S);

  // 5. edge gather + bias scatter, then attn_proj
  gather_scatter_kernel<<<EG_EDGES * H_HEADS / 256, 256, 0, stream>>>(S, src, dst, pe, sc);
  attn_proj_kernel<<<EG_EDGES * ED_DIM / 256, 256, 0, stream>>>(sc, Wap, bap, eo);

  // 6. clip + softmax over j (contiguous rows)
  softmax2048_kernel<<<H_HEADS * N_NODES, 256, 0, stream>>>(S);

  // 7. wV = A @ V per head (WMMA)
  wv_kernel<<<128, 256, 0, stream>>>(S, Vm, wV);

  // 8. output projections
  gemm_wmma_kernel<<<gqkv, 128, 0, stream>>>(wV, nullptr, Wo, bo, nullptr, ho, N_NODES, HD_DIM, HD_DIM, 0);
  dim3 goe(EG_EDGES / 64, ED_DIM / 64);
  gemm_wmma_kernel<<<goe, 128, 0, stream>>>(eo, en, Woe, boe, nullptr, eo2, EG_EDGES, ED_DIM, ED_DIM, 0);

  // 9. residuals
  add2_kernel<<<(N_NODES * HD_DIM) / 256, 256, 0, stream>>>(x, ho, outx, N_NODES * HD_DIM);
  add2_kernel<<<(EG_EDGES * ED_DIM) / 256, 256, 0, stream>>>(e, eo2, e2, EG_EDGES * ED_DIM);

  // 10. second norms
  graph_norm_kernel<<<HD_DIM, 256, 0, stream>>>(outx, gn2_w, gn2_b, gn2_a, x2g, N_NODES, HD_DIM);
  layer_norm64_kernel<<<EG_EDGES / 8, 256, 0, stream>>>(e2, ln2ew, ln2eb, e2ln, EG_EDGES);

  // 11. node FFN (WMMA, gelu) + residual into outx
  dim3 gh1(N_NODES / 64, FF_DIM / 64);
  gemm_wmma_kernel<<<gh1, 128, 0, stream>>>(x2g, nullptr, Wh1, bh1, nullptr, hh, N_NODES, HD_DIM, FF_DIM, 1);
  dim3 gh2(N_NODES / 64, HD_DIM / 64);
  gemm_wmma_kernel<<<gh2, 128, 0, stream>>>(hh, nullptr, Wh2, bh2, outx, outx, N_NODES, FF_DIM, HD_DIM, 0);

  // 12. fused edge FFN + residual into oute
  e_ffn_kernel<<<EG_EDGES / 64, 128, 0, stream>>>(e2ln, e2, We1, be1, We2, be2, oute);
}
